// LinearBase_fp8_51745765982772
// MI455X (gfx1250) — 8-bit matrix operand order repaired and checked by reading, compiled, not run on hardware
//
#include <hip/hip_runtime.h>
#include <hip/hip_bf16.h>


typedef __attribute__((ext_vector_type(16))) int   v16i;
typedef __attribute__((ext_vector_type(8)))  float v8f;
typedef __attribute__((ext_vector_type(4)))  int   v4i;
typedef __attribute__((ext_vector_type(2)))  int   v2i;

static constexpr int kT = 8192;
static constexpr int kI = 4096;
static constexpr int kO = 6144;
static constexpr int kBS = 128;
static constexpr int kKB = kI / kBS;
static constexpr float kFP8MAX = 448.0f;


__device__ __forceinline__ unsigned int f32_to_e4m3_sw(float f) {
  unsigned int u = __float_as_uint(f);
  unsigned int s = (u >> 24) & 0x80u;
  float a = __uint_as_float(u & 0x7fffffffu);
  if (a != a) return s | 0x7fu;
  if (a >= 448.0f) return s | 0x7eu;
  if (a < 0x1p-10f) return s;
  if (a < 0x1p-6f) {
    unsigned int q = (unsigned int)rintf(a * 512.0f);
    if (q > 7u) return s | 0x08u;
    return s | q;
  }
  unsigned int b = __float_as_uint(a);
  b += 0x0007FFFFu + ((b >> 20) & 1u);
  int e = (int)((b >> 23) & 0xFFu) - 127;
  if (e > 8) return s | 0x7eu;
  unsigned int m = (b >> 20) & 7u;
  return s | ((unsigned int)(e + 7) << 3) | m;
}

__device__ __forceinline__ int pack4_fp8(float a, float b, float c, float d) {
#if defined(__gfx1250__) && __has_builtin(__builtin_amdgcn_cvt_pk_fp8_f32)
  int v = __builtin_amdgcn_cvt_pk_fp8_f32(a, b, 0, false);
  v = __builtin_amdgcn_cvt_pk_fp8_f32(c, d, v, true);
  return v;
#else
  return (int)(f32_to_e4m3_sw(a) | (f32_to_e4m3_sw(b) << 8) |
               (f32_to_e4m3_sw(c) << 16) | (f32_to_e4m3_sw(d) << 24));
#endif
}

__global__ __launch_bounds__(256) void quant_x_kernel(
    const float* __restrict__ x, unsigned char* __restrict__ xq,
    float* __restrict__ xs) {
  const int t = blockIdx.x;
  const int tid = threadIdx.x;
  const float4* rp = (const float4*)(x + (size_t)t * kI);

  float4 v[4];
  float am = 0.0f;
#pragma unroll
  for (int j = 0; j < 4; ++j) {
    v[j] = rp[tid + 256 * j];
    am = fmaxf(am, fmaxf(fmaxf(fabsf(v[j].x), fabsf(v[j].y)),
                         fmaxf(fabsf(v[j].z), fabsf(v[j].w))));
  }

  __shared__ float red[256];
  red[tid] = am;
  __syncthreads();
#pragma unroll
  for (int s = 128; s > 0; s >>= 1) {
    if (tid < s) red[tid] = fmaxf(red[tid], red[tid + s]);
    __syncthreads();
  }
  const float amax = fmaxf(red[0], 1e-12f);
  const float inv = kFP8MAX / amax;

  int* op = (int*)(xq + (size_t)t * kI);
#pragma unroll
  for (int j = 0; j < 4; ++j)
    op[tid + 256 * j] =
        pack4_fp8(v[j].x * inv, v[j].y * inv, v[j].z * inv, v[j].w * inv);

  if (tid == 0) xs[t] = amax * (1.0f / kFP8MAX);
}

__global__ __launch_bounds__(256) void quant_w_kernel(
    const float* __restrict__ w, unsigned char* __restrict__ wq) {
  const size_t base = (size_t)blockIdx.x * 4096;
  const int tid = threadIdx.x;
  const float4* rp = (const float4*)(w + base);
  int* op = (int*)(wq + base);
#pragma unroll
  for (int j = 0; j < 4; ++j) {
    float4 f = rp[tid + 256 * j];
    op[tid + 256 * j] = pack4_fp8(f.x, f.y, f.z, f.w);
  }
}

__global__ __launch_bounds__(256) void gemm_fp8_kernel(
    const unsigned char* __restrict__ xq, const unsigned char* __restrict__ wq,
    const float* __restrict__ xs, const float* __restrict__ wsc,
    float* __restrict__ out) {
  const int wave = threadIdx.x >> 5;
  const int lane = threadIdx.x & 31;
  const int hi = lane >> 4;
  const int l16 = lane & 15;
  const int wm = wave >> 2;
  const int wn = wave & 3;
  const int m0 = blockIdx.x * 128 + wm * 64;
  const int o0 = blockIdx.y * 128 + wn * 32;

  const float* __restrict__ srow = wsc + (size_t)(o0 >> 7) * kKB;

  v8f acc[4][2];
#pragma unroll
  for (int mi = 0; mi < 4; ++mi)
#pragma unroll
    for (int ni = 0; ni < 2; ++ni) acc[mi][ni] = {};

  const unsigned char* aptr[4];
#pragma unroll
  for (int mi = 0; mi < 4; ++mi)
    aptr[mi] = xq + (size_t)(m0 + mi * 16 + l16) * kI + hi * 16;
  const unsigned char* bptr[2];
#pragma unroll
  for (int ni = 0; ni < 2; ++ni)
    bptr[ni] = wq + (size_t)(o0 + ni * 16 + l16) * kI + hi * 16;

  for (int kb = 0; kb < kKB; ++kb) {
    const int k0 = kb * kBS;
    const float s = srow[kb];

    union { v16i v; v2i d[8]; } A[4];
#pragma unroll
    for (int mi = 0; mi < 4; ++mi)
#pragma unroll
      for (int j = 0; j < 8; ++j)
        A[mi].d[j] = *(const v2i*)(aptr[mi] + k0 + (j >> 1) * 32 + (j & 1) * 8);

    union { v16i v; v4i q[4]; } B[2];
#pragma unroll
    for (int ni = 0; ni < 2; ++ni)
#pragma unroll
      for (int j = 0; j < 4; ++j)
        B[ni].q[j] = *(const v4i*)(bptr[ni] + k0 + j * 32);

#pragma unroll
    for (int mi = 0; mi < 4; ++mi)
#pragma unroll
      for (int ni = 0; ni < 2; ++ni) {
        v8f z = {};
        v8f d = __builtin_amdgcn_wmma_f32_16x16x128_fp8_fp8(
            A[mi].v, B[ni].v, (short)0, z, false, false);
#pragma unroll
        for (int e = 0; e < 8; ++e)
          acc[mi][ni][e] = fmaf(d[e], s, acc[mi][ni][e]);
      }
  }

#pragma unroll
  for (int mi = 0; mi < 4; ++mi) {
    const int mrow = m0 + mi * 16 + hi * 8;
    float sc[8];
    *(float4*)&sc[0] = *(const float4*)(xs + mrow);
    *(float4*)&sc[4] = *(const float4*)(xs + mrow + 4);
#pragma unroll
    for (int ni = 0; ni < 2; ++ni) {
      const int ocol = o0 + ni * 16 + l16;
#pragma unroll
      for (int r = 0; r < 8; ++r)
        out[(size_t)(mrow + r) * kO + ocol] = acc[mi][ni][r] * sc[r];
    }
  }
}


extern "C" void kernel_launch(void* const* d_in, const int* in_sizes, int n_in,
                              void* d_out, int out_size, void* d_ws,
                              size_t ws_size, hipStream_t stream) {
  const float* x = (const float*)d_in[0];
  const float* w = (const float*)d_in[1];
  const float* wsc = (const float*)d_in[2];
  float* out = (float*)d_out;

  unsigned char* xq = (unsigned char*)d_ws;
  unsigned char* wq = xq + (size_t)kT * kI;
  float* xs = (float*)(wq + (size_t)kO * kI);

  quant_x_kernel<<<kT, 256, 0, stream>>>(x, xq, xs);
  quant_w_kernel<<<(int)(((size_t)kO * kI) / 4096), 256, 0, stream>>>(w, wq);

  dim3 grid(kT / 128, kO / 128);
  gemm_fp8_kernel<<<grid, 256, 0, stream>>>(xq, wq, xs, wsc, out);
}
